// ATT_ONE2MANY_1_54322746360557
// MI455X (gfx1250) — matrix operand order repaired and checked by reading, not compiled, not run on hardware
//
#include <hip/hip_runtime.h>

// ---------------------------------------------------------------------------
// Shapes (fixed by the reference): one [B,D], many [B,T,D], W [D,D]
// out = (result [B,D], att_weights [B,T]) concatenated, fp32.
// ---------------------------------------------------------------------------
#define B_   32
#define T_   2048
#define D_   1024

// GEMM tiling for kernel 1
#define BM        64                 // rows of `many` per workgroup
#define BN        128                // e-columns per outer iteration
#define BK        32                 // K step (= WMMA bf16 K)
#define NKS       (D_ / BK)          // 32 k-steps per e-tile
#define SA        (D_ + 8)           // LDS A row stride (bf16 elems), 16B-aligned rows
#define SW        (BK + 8)           // LDS Wt row stride (bf16 elems), 16B-aligned rows
#define NTHREADS  256                // 8 wave32

typedef __attribute__((ext_vector_type(16))) __bf16 v16bf;
typedef __attribute__((ext_vector_type(8)))  float  v8f;

union ABfrag { v16bf v; uint4 q[2]; };

// fp32 -> bf16, round-to-nearest-even (self-contained)
static __device__ __forceinline__ unsigned short f2bf(float f) {
  union { float f; unsigned int u; } x; x.f = f;
  unsigned int r = x.u + 0x7FFFu + ((x.u >> 16) & 1u);
  return (unsigned short)(r >> 16);
}

// Branch-free tanh: no EXEC divergence, native v_exp_f32 + v_rcp_f32.
static __device__ __forceinline__ float fast_tanh(float x) {
  float xc = fminf(fmaxf(x, -15.0f), 15.0f);
  float e2 = __expf(2.0f * xc);
  return __fdividef(e2 - 1.0f, e2 + 1.0f);
}

// ---------------------------------------------------------------------------
// Kernel 0: one-time W [k][e] fp32 -> Wt [e][k] bf16 (transposed, in d_ws).
// LDS-tiled transpose, coalesced on both sides. 4MB read / 2MB write.
// ---------------------------------------------------------------------------
__global__ void __launch_bounds__(256)
w_transpose_bf16_kernel(const float* __restrict__ W,
                        unsigned short* __restrict__ Wt)
{
  __shared__ unsigned short tile[32][34];
  const int tx = threadIdx.x & 31;
  const int ty = threadIdx.x >> 5;         // 0..7
  const int k0 = blockIdx.x * 32;
  const int e0 = blockIdx.y * 32;
  #pragma unroll
  for (int i = 0; i < 4; ++i) {
    const int k = ty + i * 8;
    tile[k][tx] = f2bf(W[(size_t)(k0 + k) * D_ + e0 + tx]);
  }
  __syncthreads();
  #pragma unroll
  for (int i = 0; i < 4; ++i) {
    const int e = ty + i * 8;
    Wt[(size_t)(e0 + e) * D_ + k0 + tx] = tile[tx][e];
  }
}

// ---------------------------------------------------------------------------
// Kernel 1: logits[b,t] = sum_e tanh( sum_k many[b,t,k]*W[k,e] ) * one[b,e]
// A tile (64x1024) staged once to LDS as bf16 (132KB of the 320KB WGP LDS);
// Wt chunks double-buffered with register prefetch: one barrier per k-step.
// ---------------------------------------------------------------------------
__global__ void __launch_bounds__(NTHREADS)
att_logits_kernel(const float* __restrict__ many,
                  const unsigned short* __restrict__ Wt,   // bf16 [e][k]
                  const float* __restrict__ one,
                  float* __restrict__ logits_out)          // -> att region of d_out
{
  extern __shared__ char smem[];
  __bf16* As  = (__bf16*)smem;                                          // [BM][SA]
  __bf16* Wb0 = (__bf16*)(smem + (size_t)BM * SA * 2);                  // [BN][SW]
  __bf16* Wb1 = Wb0 + (size_t)BN * SW;                                  // [BN][SW]
  float*  Ll  = (float*)(smem + ((size_t)BM * SA + 2u * BN * SW) * 2);  // [BM]

  const int tid  = threadIdx.x;
  const int wave = tid >> 5;
  const int lane = tid & 31;
  const int wm   = wave & 1;          // 2 wave-rows (M)
  const int wn   = wave >> 1;         // 4 wave-cols (N)
  const int lh   = lane >> 4;         // WMMA lane half
  const int ln   = lane & 15;

  const int b  = blockIdx.x >> 5;                // T_/BM = 32 tiles per batch
  const int t0 = (blockIdx.x & 31) * BM;

  if (tid < BM) Ll[tid] = 0.0f;

  // ---- Stage A tile: many[b, t0:t0+64, :] fp32 -> bf16 LDS (one time) ----
  {
    const float4* Ab = (const float4*)(many + ((size_t)b * T_ + t0) * D_);
    unsigned long long* AsU = (unsigned long long*)As;
    for (int f = tid; f < BM * (D_ / 4); f += NTHREADS) {
      const int row = f >> 8;
      const int c4  = f & 255;
      float4 x = Ab[row * (D_ / 4) + c4];
      unsigned long long p =
            (unsigned long long)f2bf(x.x)
          | ((unsigned long long)f2bf(x.y) << 16)
          | ((unsigned long long)f2bf(x.z) << 32)
          | ((unsigned long long)f2bf(x.w) << 48);
      AsU[((size_t)row * SA + c4 * 4) >> 2] = p;
    }
  }

  float rp[2][8];
  #pragma unroll
  for (int mt = 0; mt < 2; ++mt)
    #pragma unroll
    for (int j = 0; j < 8; ++j) rp[mt][j] = 0.0f;

  // staging helpers: 512 uint4 chunks (128 e-rows x 4), 2 per thread
  const int sc_e  = (tid + 0)   >> 2;        // e row for chunk 0 (tid<512 always)
  const int sc_k0 = ((tid + 0)  & 3) * 8;    // k offset (bf16 elems)
  const int sc_e1  = (tid + NTHREADS) >> 2;
  const int sc_k1 = ((tid + NTHREADS) & 3) * 8;

  for (int et = 0; et < D_ / BN; ++et) {
    const int e0 = et * BN;

    v8f acc[2][2];
    #pragma unroll
    for (int mt = 0; mt < 2; ++mt)
      #pragma unroll
      for (int nt = 0; nt < 2; ++nt)
        acc[mt][nt] = (v8f){0.f,0.f,0.f,0.f,0.f,0.f,0.f,0.f};

    uint4 r0, r1;
    // prologue: load & store chunk ks=0 into buffer 0
    r0 = *(const uint4*)(Wt + (size_t)(e0 + sc_e)  * D_ + sc_k0);
    r1 = *(const uint4*)(Wt + (size_t)(e0 + sc_e1) * D_ + sc_k1);
    *(uint4*)&Wb0[sc_e  * SW + sc_k0] = r0;
    *(uint4*)&Wb0[sc_e1 * SW + sc_k1] = r1;

    for (int ks = 0; ks < NKS; ++ks) {
      const int k0 = ks * BK;
      __bf16* WsC = (ks & 1) ? Wb1 : Wb0;
      __bf16* WsN = (ks & 1) ? Wb0 : Wb1;

      // prefetch next chunk into registers (global latency hides under WMMA)
      if (ks + 1 < NKS) {
        const int kn = (ks + 1) * BK;
        r0 = *(const uint4*)(Wt + (size_t)(e0 + sc_e)  * D_ + kn + sc_k0);
        r1 = *(const uint4*)(Wt + (size_t)(e0 + sc_e1) * D_ + kn + sc_k1);
      }

      __syncthreads();  // cur buffer stores visible; prior reads of next buffer done

      // A 16x32 bf16 fragments: lane<16 -> M=lane, K=0..7 & 16..23 ; else K=8..15 & 24..31
      ABfrag a[2];
      #pragma unroll
      for (int mt = 0; mt < 2; ++mt) {
        const __bf16* p = &As[(size_t)(wm * 32 + mt * 16 + ln) * SA + k0 + lh * 8];
        a[mt].q[0] = *(const uint4*)(p);
        a[mt].q[1] = *(const uint4*)(p + 16);
      }
      // B 32x16 bf16 fragments: lane<16 -> N=lane, K=0..7 & 16..23 ; else K=8..15 & 24..31
      ABfrag bfr[2];
      #pragma unroll
      for (int nt = 0; nt < 2; ++nt) {
        const __bf16* p = &WsC[(size_t)(wn * 32 + nt * 16 + ln) * SW + lh * 8];
        bfr[nt].q[0] = *(const uint4*)(p);
        bfr[nt].q[1] = *(const uint4*)(p + 16);
      }

      #pragma unroll
      for (int mt = 0; mt < 2; ++mt)
        #pragma unroll
        for (int nt = 0; nt < 2; ++nt)
          acc[mt][nt] = __builtin_amdgcn_wmma_f32_16x16x32_bf16(
              false, a[mt].v, false, bfr[nt].v,
              (short)0, acc[mt][nt], false, false);

      if (ks + 1 < NKS) {
        *(uint4*)&WsN[sc_e  * SW + sc_k0] = r0;
        *(uint4*)&WsN[sc_e1 * SW + sc_k1] = r1;
      }
    }

    // ---- epilogue: tanh + dot with one[b,...], branch-free ----
    #pragma unroll
    for (int nt = 0; nt < 2; ++nt) {
      const float ov = one[(size_t)b * D_ + e0 + wn * 32 + nt * 16 + ln];
      #pragma unroll
      for (int mt = 0; mt < 2; ++mt)
        #pragma unroll
        for (int j = 0; j < 8; ++j)
          rp[mt][j] += fast_tanh(acc[mt][nt][j]) * ov;
    }
  }

  __syncthreads();
  #pragma unroll
  for (int mt = 0; mt < 2; ++mt)
    #pragma unroll
    for (int j = 0; j < 8; ++j)
      atomicAdd(&Ll[wm * 32 + mt * 16 + lh * 8 + j], rp[mt][j]);  // ds_add_f32
  __syncthreads();

  if (tid < BM)
    logits_out[(size_t)b * T_ + t0 + tid] = Ll[tid];
}

// ---------------------------------------------------------------------------
// Kernel 2: in-place softmax over T per batch row (Keras epsilon in denom)
// ---------------------------------------------------------------------------
__global__ void __launch_bounds__(256)
att_softmax_kernel(float* __restrict__ att)
{
  __shared__ float red[256];
  const int b = blockIdx.x, tid = threadIdx.x;
  float v[8];
  float m = -3.402823466e38f;
  #pragma unroll
  for (int i = 0; i < 8; ++i) {
    v[i] = att[(size_t)b * T_ + tid + i * 256];
    m = fmaxf(m, v[i]);
  }
  red[tid] = m; __syncthreads();
  for (int s = 128; s > 0; s >>= 1) {
    if (tid < s) red[tid] = fmaxf(red[tid], red[tid + s]);
    __syncthreads();
  }
  m = red[0]; __syncthreads();

  float sum = 0.0f;
  #pragma unroll
  for (int i = 0; i < 8; ++i) { v[i] = __expf(v[i] - m); sum += v[i]; }
  red[tid] = sum; __syncthreads();
  for (int s = 128; s > 0; s >>= 1) {
    if (tid < s) red[tid] += red[tid + s];
    __syncthreads();
  }
  const float inv = 1.0f / (red[0] + 1e-7f);
  #pragma unroll
  for (int i = 0; i < 8; ++i)
    att[(size_t)b * T_ + tid + i * 256] = v[i] * inv;
}

// ---------------------------------------------------------------------------
// Kernel 3: partial[b,tc,d] = sum_{t in chunk} many[b,t,d] * att[b,t]
// Weights staged to LDS once per chunk; float4 streaming of `many`.
// ---------------------------------------------------------------------------
__global__ void __launch_bounds__(256)
att_wsum_partial_kernel(const float* __restrict__ many,
                        const float* __restrict__ att,
                        float* __restrict__ partial)
{
  __shared__ float wsh[256];
  const int b  = blockIdx.x >> 3;
  const int tc = blockIdx.x & 7;
  const int tid = threadIdx.x;
  wsh[tid] = att[(size_t)b * T_ + tc * 256 + tid];
  __syncthreads();
  const float* mb = many + (size_t)b * T_ * D_ + (size_t)(tc * 256) * D_;
  float4 acc = {0.f, 0.f, 0.f, 0.f};
  for (int t = 0; t < 256; ++t) {
    const float w = wsh[t];
    float4 m4 = *(const float4*)(mb + (size_t)t * D_ + tid * 4);
    acc.x += w * m4.x; acc.y += w * m4.y; acc.z += w * m4.z; acc.w += w * m4.w;
  }
  *(float4*)(partial + (size_t)blockIdx.x * D_ + tid * 4) = acc;
}

// ---------------------------------------------------------------------------
// Kernel 4: result[b,d] = sum_{tc} partial[b,tc,d]
// ---------------------------------------------------------------------------
__global__ void __launch_bounds__(256)
att_wsum_reduce_kernel(const float* __restrict__ partial,
                       float* __restrict__ result)
{
  const int b = blockIdx.x, tid = threadIdx.x;
  float4 acc = {0.f, 0.f, 0.f, 0.f};
  #pragma unroll
  for (int i = 0; i < 8; ++i) {
    float4 p = *(const float4*)(partial + (size_t)(b * 8 + i) * D_ + tid * 4);
    acc.x += p.x; acc.y += p.y; acc.z += p.z; acc.w += p.w;
  }
  *(float4*)(result + (size_t)b * D_ + tid * 4) = acc;
}

// ---------------------------------------------------------------------------
extern "C" void kernel_launch(void* const* d_in, const int* in_sizes, int n_in,
                              void* d_out, int out_size, void* d_ws, size_t ws_size,
                              hipStream_t stream) {
  (void)in_sizes; (void)n_in; (void)out_size; (void)ws_size;
  const float* one  = (const float*)d_in[0];   // [B,D]
  const float* many = (const float*)d_in[1];   // [B,T,D]
  const float* W    = (const float*)d_in[2];   // [D,D]

  float* result = (float*)d_out;               // [B,D]
  float* att    = (float*)d_out + B_ * D_;     // [B,T] logits -> softmax in place

  unsigned short* Wt = (unsigned short*)d_ws;                          // bf16 [D][D] = 2MB
  float* partial = (float*)((char*)d_ws + (size_t)D_ * D_ * 2);        // [B*8][D]  = 1MB

  const size_t smem = ((size_t)BM * SA + 2u * BN * SW) * 2 + BM * sizeof(float);

  w_transpose_bf16_kernel<<<dim3(D_ / 32, D_ / 32), dim3(256), 0, stream>>>(W, Wt);
  att_logits_kernel<<<dim3(B_ * (T_ / BM)), dim3(NTHREADS), smem, stream>>>(many, Wt, one, att);
  att_softmax_kernel<<<dim3(B_), dim3(256), 0, stream>>>(att);
  att_wsum_partial_kernel<<<dim3(B_ * 8), dim3(256), 0, stream>>>(many, att, partial);
  att_wsum_reduce_kernel<<<dim3(B_), dim3(256), 0, stream>>>(partial, result);
}
